// MultiLinearResSSM1_17557826306694
// MI455X (gfx1250) — matrix operand order repaired and checked by reading, not compiled, not run on hardware
//
#include <hip/hip_runtime.h>
#include <hip/hip_bf16.h>

#define SD 140      // state dim
#define AD 40       // action dim
#define SO 10       // state order
#define AO 10       // action order
#define NBATCH 256
#define HZ 1024

#define NPAD 144            // N padded to 9 tiles of 16
#define LAGS 160            // per-lag K stride (bf16 elems), multiple of 32
#define KROW (SO*LAGS)      // 1600
#define KCH  (KROW/32)      // 50 k-chunks for scan
#define UPAD 64             // per-lag action stride padded
#define KU   (AO*UPAD)      // 640
#define KUCH (KU/32)        // 20 k-chunks for exo

typedef __attribute__((ext_vector_type(16))) __bf16 v16bf;
typedef __attribute__((ext_vector_type(8)))  float  v8f;
typedef __attribute__((ext_vector_type(4)))  unsigned int v4u;

union Frag {
  v16bf v;
  v4u   q[2];
  unsigned short h[16];
};

__device__ __forceinline__ unsigned short f2bf(float f) {
  unsigned int u = __float_as_uint(f);
  unsigned int r = u + 0x7FFFu + ((u >> 16) & 1u);   // round-to-nearest-even
  return (unsigned short)(r >> 16);
}

// ---- prep: A [10,140,140] f32 -> At[n][o*160+x] bf16 (N-major, zero padded)
__global__ void prep_At(const float* __restrict__ A, unsigned short* __restrict__ At) {
  int idx = blockIdx.x * blockDim.x + threadIdx.x;
  if (idx >= NPAD * KROW) return;
  int n = idx / KROW;
  int k = idx % KROW;
  int o = k / LAGS;
  int x = k % LAGS;
  unsigned short r = 0;
  if (x < SD && n < SD) r = f2bf(A[((size_t)o * SD + x) * SD + n]);
  At[idx] = r;
}

// ---- prep: B [10,40,140] f32 -> Bt[n][o*64+a] bf16 (N-major, zero padded)
__global__ void prep_Bt(const float* __restrict__ Bm, unsigned short* __restrict__ Bt) {
  int idx = blockIdx.x * blockDim.x + threadIdx.x;
  if (idx >= NPAD * KU) return;
  int n = idx / KU;
  int k = idx % KU;
  int o = k / UPAD;
  int a = k % UPAD;
  unsigned short r = 0;
  if (a < AD && n < SD) r = f2bf(Bm[((size_t)o * AD + a) * SD + n]);
  Bt[idx] = r;
}

// ---- exo[b][t][n] = sum_o u_cat[b][t+o] @ B[o] ; fully parallel WMMA GEMM
__global__ __launch_bounds__(256)
void exo_kernel(const float* __restrict__ u0, const float* __restrict__ us,
                const unsigned short* __restrict__ Bt, float* __restrict__ exo) {
  __shared__ unsigned short uw[73 * UPAD];   // sliding u window, bf16, zero padded
  const int b  = blockIdx.y;
  const int t0 = blockIdx.x * 64;

  for (int i = threadIdx.x; i < 73 * UPAD; i += blockDim.x) {
    int tr = i / UPAD, a = i % UPAD;
    int j = t0 + tr;                          // u_cat time index (< 1033)
    float val = 0.f;
    if (a < AD) {
      val = (j < AO - 1) ? u0[((size_t)b * (AO - 1) + j) * AD + a]
                         : us[((size_t)b * HZ + (j - (AO - 1))) * AD + a];
    }
    uw[i] = f2bf(val);
  }
  __syncthreads();

  const int wave = threadIdx.x >> 5;
  const int lane = threadIdx.x & 31;
  const int hi   = (lane < 16) ? 0 : 1;
  const int l16  = lane & 15;

  for (int job = wave; job < 36; job += 8) {   // 4 M-tiles x 9 N-tiles
    const int mt = job / 9;
    const int nt = job % 9;
    const int n  = nt * 16 + l16;
    const int mrow = mt * 16 + l16;
    const unsigned short* btrow = Bt + (size_t)n * KU;
    v8f acc = {0.f,0.f,0.f,0.f,0.f,0.f,0.f,0.f};
    #pragma unroll
    for (int kk = 0; kk < KUCH; ++kk) {
      const int o = kk >> 1, c = kk & 1;
      Frag af, bf;
      const unsigned short* ap = &uw[(mrow + o) * UPAD + c * 32 + hi * 8];
      af.q[0] = *(const v4u*)(ap);
      af.q[1] = *(const v4u*)(ap + 16);
      const unsigned short* bp = btrow + kk * 32 + hi * 8;
      bf.q[0] = *(const v4u*)(bp);
      bf.q[1] = *(const v4u*)(bp + 16);
      acc = __builtin_amdgcn_wmma_f32_16x16x32_bf16(false, af.v, false, bf.v,
                                                    (short)0, acc, false, false);
    }
    #pragma unroll
    for (int v = 0; v < 8; ++v) {
      int t = t0 + mt * 16 + v + hi * 8;       // C-layout: M = v (+8 for hi half)
      exo[((size_t)b * HZ + t) * NPAD + n] = acc[v];
    }
  }
}

// ---- sequential scan: 16 WGs, 16 batch rows each; At resident in VGPRs
__global__ __launch_bounds__(864)
void scan_kernel(const float* __restrict__ x0,
                 const unsigned short* __restrict__ At,
                 const float* __restrict__ exo,
                 float* __restrict__ xs) {
  __shared__ unsigned short buf[16 * KROW];    // state ring, bf16, 51.2 KB
  __shared__ float red[2][9][32][8];           // K-partial reduction, 18.4 KB

  const int b0   = blockIdx.x * 16;
  const int tid  = threadIdx.x;
  const int wave = tid >> 5;
  const int lane = tid & 31;
  const int hi   = (lane < 16) ? 0 : 1;
  const int l16  = lane & 15;
  const int nt   = wave % 9;                   // N tile
  const int kp   = wave / 9;                   // K partition 0..2
  const int n    = nt * 16 + l16;
  const int mrow0 = hi * 8;

  for (int i = tid; i < 16 * KROW; i += blockDim.x) buf[i] = 0;
  __syncthreads();
  for (int i = tid; i < 16 * SO * SD; i += blockDim.x) {
    int m = i / (SO * SD);
    int r = i % (SO * SD);
    int o = r / SD, x = r % SD;
    buf[m * KROW + o * LAGS + x] = f2bf(x0[((size_t)(b0 + m) * SO + o) * SD + x]);
  }

  // preload this wave's slice of At into registers (B-fragments)
  const int kbase = kp * 17;
  Frag bfr[17];
  #pragma unroll
  for (int i = 0; i < 17; ++i) {
    int kk = kbase + i;
    if (kk < KCH) {
      const unsigned short* bp = At + (size_t)n * KROW + kk * 32 + hi * 8;
      bfr[i].q[0] = *(const v4u*)(bp);
      bfr[i].q[1] = *(const v4u*)(bp + 16);
    } else {
      v4u z = {0u, 0u, 0u, 0u};
      bfr[i].q[0] = z;                         // zero frag: contributes exactly 0
      bfr[i].q[1] = z;
    }
  }

  // f32 residual state (owner waves): C-layout tile of x_{t-1}
  v8f xprev;
  #pragma unroll
  for (int v = 0; v < 8; ++v) {
    int m = v + mrow0;
    xprev[v] = (n < SD) ? x0[((size_t)(b0 + m) * SO + (SO - 1)) * SD + n] : 0.f;
  }
  __syncthreads();

  const char*  bufBytes = (const char*)buf;
  // A-frag byte offset is affine in kk: o*LAGS + c*32 = 32*kk (LAGS = 5*32).
  // Per-wave uniform part; frag i adds i*64 bytes; ring adds ph*320 with one wrap.
  const int    fragBase = l16 * (KROW * 2) + kbase * 64 + hi * 16;
  const float* exoPtr   = exo + (size_t)(b0 + mrow0) * HZ * NPAD + n;
  float*       xsPtr    = xs  + (size_t)(b0 + mrow0) * HZ * SD + n;
  float*       redW     = (kp != 0) ? &red[kp - 1][nt][lane][0] : nullptr;
  const float* r0       = &red[0][nt][lane][0];
  const float* r1       = &red[1][nt][lane][0];

  auto step = [&](int ph, int t) {
    // prefetch exo for this step (owner waves)
    v8f exoR = {0.f,0.f,0.f,0.f,0.f,0.f,0.f,0.f};
    if (kp == 0) {
      #pragma unroll
      for (int v = 0; v < 8; ++v)
        exoR[v] = exoPtr[((size_t)v * HZ + t) * NPAD];
    }
    // wmma over this wave's resident K-slice
    const int base = fragBase + ph * (LAGS * 2);
    v8f acc = {0.f,0.f,0.f,0.f,0.f,0.f,0.f,0.f};
    #pragma unroll
    for (int i = 0; i < 17; ++i) {
      int p = base + i * 64;                   // ring rotation, branch-free wrap
      if (p - l16 * (KROW * 2) >= KROW * 2) p -= KROW * 2;
      Frag af;
      const char* ap = bufBytes + p;
      af.q[0] = *(const v4u*)(ap);
      af.q[1] = *(const v4u*)(ap + 32);
      acc = __builtin_amdgcn_wmma_f32_16x16x32_bf16(false, af.v, false, bfr[i].v,
                                                    (short)0, acc, false, false);
    }
    if (kp != 0) {                             // partners publish partials
      #pragma unroll
      for (int v = 0; v < 8; ++v) redW[v] = acc[v];
    }
    __syncthreads();
    if (kp == 0) {                             // owners reduce + finalize
      #pragma unroll
      for (int v = 0; v < 8; ++v) {
        float xv = acc[v] + r0[v] + r1[v] + exoR[v] + xprev[v];
        xprev[v] = xv;
        if (n < SD) {
          xsPtr[((size_t)v * HZ + t) * SD] = xv;
          buf[(mrow0 + v) * KROW + ph * LAGS + n] = f2bf(xv);  // newest -> slot ph
        }
      }
    }
    __syncthreads();
  };

  for (int tt = 0; tt < 1020; tt += 10) {
    #pragma unroll
    for (int ph = 0; ph < 10; ++ph) step(ph, tt + ph);   // ph constant-folds
  }
  for (int t = 1020; t < HZ; ++t) step(t - 1020, t);     // 4-step tail
}

extern "C" void kernel_launch(void* const* d_in, const int* in_sizes, int n_in,
                              void* d_out, int out_size, void* d_ws, size_t ws_size,
                              hipStream_t stream) {
  const float* x0 = (const float*)d_in[0];
  const float* u0 = (const float*)d_in[1];
  const float* us = (const float*)d_in[2];
  const float* A  = (const float*)d_in[3];
  const float* Bm = (const float*)d_in[4];
  float* xs = (float*)d_out;

  char* ws = (char*)d_ws;
  unsigned short* At = (unsigned short*)ws;                    // 450 KB
  unsigned short* Bt = (unsigned short*)(ws + (512u << 10));   // 180 KB
  float*          exo = (float*)(ws + (1u << 20));             // 151 MB

  hipLaunchKernelGGL(prep_At, dim3((NPAD * KROW + 255) / 256), dim3(256), 0, stream, A, At);
  hipLaunchKernelGGL(prep_Bt, dim3((NPAD * KU + 255) / 256), dim3(256), 0, stream, Bm, Bt);
  hipLaunchKernelGGL(exo_kernel, dim3(HZ / 64, NBATCH), dim3(256), 0, stream, u0, us, Bt, exo);
  hipLaunchKernelGGL(scan_kernel, dim3(NBATCH / 16), dim3(864), 0, stream, x0, At, exo, xs);
}
